// RNN_71262097375319
// MI455X (gfx1250) — matrix operand order repaired and checked by reading, not compiled, not run on hardware
//
#include <hip/hip_runtime.h>
#include <hip/hip_bf16.h>
#include <math.h>

typedef __attribute__((ext_vector_type(16))) __bf16 v16bf;
typedef __attribute__((ext_vector_type(8)))  __bf16 v8bf;
typedef __attribute__((ext_vector_type(4)))  __bf16 v4bf;
typedef __attribute__((ext_vector_type(8)))  float  v8f;

#define B_ 64
#define T_ 1024
#define E_ 512
#define H_ 1024
#define K_ (H_ + E_)       // 1536: fused [h | x] K dimension
#define KCH (K_ / 32)      // 48 K-chunks (0..31 = H side, 32..47 = E side)
#define NBLK 64            // persistent blocks (one per WGP)
#define COLS 16            // H columns owned per block

#define LDA_S 40           // A tile [m][k]: 32 K + 8 pad (bf16 elems)
#define LW_S  1544         // persistent weight [col][K]: 1536 + 8 pad

// ---- fragment loaders: contiguous k order (k = 16*(lane/16) + s) for A and B ----
__device__ __forceinline__ v16bf load_a_frag(const __bf16* lA, int m0, int lane) {
  const int g = lane >> 4, r = lane & 15;
  const __bf16* p = lA + (m0 + r) * LDA_S + g * 16;
  v8bf lo = *(const v8bf*)(p);
  v8bf hi = *(const v8bf*)(p + 8);
  v16bf a;
#pragma unroll
  for (int i = 0; i < 8; ++i) { a[i] = lo[i]; a[i + 8] = hi[i]; }
  return a;
}
__device__ __forceinline__ v16bf load_b_fragK(const __bf16* lB, int kofs, int lane) {
  const int g = lane >> 4, r = lane & 15;
  const __bf16* p = lB + (size_t)r * LW_S + kofs + g * 16;
  v8bf lo = *(const v8bf*)(p);
  v8bf hi = *(const v8bf*)(p + 8);
  v16bf b;
#pragma unroll
  for (int i = 0; i < 8; ++i) { b[i] = lo[i]; b[i + 8] = hi[i]; }
  return b;
}

__device__ __forceinline__ v8f wmma_bf16(v16bf a, v16bf b, v8f c) {
  return __builtin_amdgcn_wmma_f32_16x16x32_bf16(false, a, false, b, (short)0, c,
                                                 false, false);
}

__device__ __forceinline__ v4bf cvt4(const float4 f) {
  v4bf o;
  o[0] = (__bf16)f.x; o[1] = (__bf16)f.y; o[2] = (__bf16)f.z; o[3] = (__bf16)f.w;
  return o;
}

__device__ __forceinline__ float sigmoidf_(float x) {
  return 1.0f / (1.0f + __expf(-x));
}

// ---- CDNA5 async memory->LDS copy (CDNA5 ISA), ASYNCcnt ----
// Copies 32 bytes per lane: two GLOBAL_LOAD_ASYNC_TO_LDS_B128 (GV mode).
// lds_dst = LDS byte offset (flat shared pointer truncated to addr[31:0]).
__device__ __forceinline__ void async_copy32B(unsigned lds_dst, const void* gsrc) {
  const unsigned long long a = (unsigned long long)gsrc;
  asm volatile(
      "global_load_async_to_lds_b128 %0, %1, off\n\t"
      "global_load_async_to_lds_b128 %0, %1, off offset:16"
      :: "v"(lds_dst), "v"(a) : "memory");
}
__device__ __forceinline__ void wait_async0() {
  asm volatile("s_wait_asynccnt 0" ::: "memory");
}

__global__ void zero_f32(float* __restrict__ p, int n) {
  int i = blockIdx.x * blockDim.x + threadIdx.x;
  if (i < n) p[i] = 0.0f;
}

// one-shot f32 -> bf16 conversion (for x), vectorized 4-wide
__global__ void cvt_f32_bf16(const float* __restrict__ src,
                             __bf16* __restrict__ dst, int n4) {
  int i = blockIdx.x * blockDim.x + threadIdx.x;
  if (i < n4) ((v4bf*)dst)[i] = cvt4(((const float4*)src)[i]);
}

// Device-wide barrier: valid because all NBLK blocks are co-resident
// (one per WGP; ~155KB LDS each forbids doubling up).
__device__ __forceinline__ void grid_barrier(unsigned* cnt, unsigned* gen) {
  __threadfence();
  __syncthreads();
  if (threadIdx.x == 0) {
    const unsigned g = __hip_atomic_load(gen, __ATOMIC_ACQUIRE,
                                         __HIP_MEMORY_SCOPE_AGENT);
    if (__hip_atomic_fetch_add(cnt, 1u, __ATOMIC_ACQ_REL,
                               __HIP_MEMORY_SCOPE_AGENT) == NBLK - 1) {
      __hip_atomic_store(cnt, 0u, __ATOMIC_RELAXED, __HIP_MEMORY_SCOPE_AGENT);
      __hip_atomic_fetch_add(gen, 1u, __ATOMIC_RELEASE,
                             __HIP_MEMORY_SCOPE_AGENT);
    } else {
      while (__hip_atomic_load(gen, __ATOMIC_ACQUIRE,
                               __HIP_MEMORY_SCOPE_AGENT) == g) {
        __builtin_amdgcn_s_sleep(2);
      }
    }
  }
  __syncthreads();
  __threadfence();
}

// ---------------------------------------------------------------------------
// Persistent GRU recurrence. 64 blocks x 128 threads (4 waves). Weights for
// this block's 16 H-columns live in LDS (bf16, K fused as [w_hh | w_ih]),
// loaded once. A tile is double-buffered; ALL per-step staging is done with
// async memory->LDS copies (hsb history and pre-converted x are both bf16).
// ---------------------------------------------------------------------------
__global__ __launch_bounds__(128) void gru_persistent(
    const __bf16* __restrict__ xb,    // [B,T,E] bf16 (pre-converted)
    const float* __restrict__ w_ih,   // [3H,E]
    const float* __restrict__ w_hh,   // [3H,H]
    const float* __restrict__ b_ih,   // [3H]
    const float* __restrict__ b_hh,   // [3H]
    float* __restrict__ h0,           // [B,H] f32 (zeroed)
    float* __restrict__ h1,           // [B,H] f32
    __bf16* __restrict__ hsb,         // [T+1,B,H] bf16, slot 0 zeroed
    unsigned* __restrict__ bar_cnt,
    unsigned* __restrict__ bar_gen) {
  __shared__ __align__(16) __bf16 wlds[3 * COLS * LW_S];   // 144.8 KB
  __shared__ __align__(16) __bf16 lA2[2][64 * LDA_S];      //  10.0 KB

  const int tid  = threadIdx.x;
  const int lane = tid & 31;
  const int wv   = tid >> 5;
  const int m0   = wv * 16;
  const int colbase = blockIdx.x * COLS;

  // ---- one-time weight load: [q][n][k] with K fused as [w_hh | w_ih] ----
  for (int c = 0; c < 3 * COLS; ++c) {
    const int q = c >> 4, n = c & (COLS - 1);
    const size_t row = (size_t)(q * H_ + colbase + n);
    for (int e = tid * 4; e < K_; e += 512) {
      const float4 f = (e < H_)
          ? *(const float4*)(w_hh + row * H_ + e)
          : *(const float4*)(w_ih + row * E_ + (e - H_));
      *(v4bf*)&wlds[(size_t)c * LW_S + e] = cvt4(f);
    }
  }
  __syncthreads();

  const int g   = lane >> 4;
  const int col = colbase + (lane & 15);
  const float br  = b_ih[col] + b_hh[col];
  const float bz  = b_ih[H_ + col] + b_hh[H_ + col];
  const float bnx = b_ih[2 * H_ + col];
  const float bnh = b_hh[2 * H_ + col];

  // A staging: 64 rows x 32 K = 2048 bf16 / 128 thr = 32B per thread
  const int ar = tid >> 1;              // 0..63 batch row
  const int ak = (tid & 1) * 16;        // 0 or 16

  const unsigned ldsA0 = (unsigned)(uintptr_t)&lA2[0][ar * LDA_S + ak];
  const unsigned ldsA1 = (unsigned)(uintptr_t)&lA2[1][ar * LDA_S + ak];

  const __bf16* wR = wlds;
  const __bf16* wZ = wlds + (size_t)COLS * LW_S;
  const __bf16* wN = wlds + (size_t)2 * COLS * LW_S;

  v8f accR, accZ, accNh, accNx;
  auto compute = [&](int c, const __bf16* buf, v8f& accN) {
    const int kofs = c * 32;
    v16bf a  = load_a_frag(buf, m0, lane);
    v16bf b0 = load_b_fragK(wR, kofs, lane);
    v16bf b1 = load_b_fragK(wZ, kofs, lane);
    v16bf b2 = load_b_fragK(wN, kofs, lane);
    accR = wmma_bf16(a, b0, accR);
    accZ = wmma_bf16(a, b1, accZ);
    accN = wmma_bf16(a, b2, accN);
  };

#pragma unroll 1
  for (int t = 0; t < T_; ++t) {
    const float* __restrict__ hp = (t & 1) ? h1 : h0;
    float* __restrict__       hn = (t & 1) ? h0 : h1;
    const __bf16* __restrict__ hrow = hsb + (size_t)t * (B_ * H_) +
                                      (size_t)ar * H_ + ak;          // h_{t-1}
    const __bf16* __restrict__ xrow = xb + ((size_t)ar * T_ + t) * E_ + ak;

    accR = v8f{}; accZ = v8f{}; accNh = v8f{}; accNx = v8f{};

    // prologue: chunk 0 (H side) -> buf0
    async_copy32B(ldsA0, hrow);
    wait_async0();
    __syncthreads();

    // H side: chunks 0..30, staging chunk c+1 while computing chunk c
#pragma unroll 2
    for (int c = 0; c < 31; ++c) {
      async_copy32B((c & 1) ? ldsA0 : ldsA1, hrow + (c + 1) * 32);
      compute(c, lA2[c & 1], accNh);
      wait_async0();
      __syncthreads();
    }
    // boundary: stage first E chunk (32) into buf0, compute chunk 31
    async_copy32B(ldsA0, xrow);
    compute(31, lA2[1], accNh);
    wait_async0();
    __syncthreads();
    // E side: chunks 32..46
#pragma unroll 2
    for (int c = 32; c < 47; ++c) {
      async_copy32B((c & 1) ? ldsA0 : ldsA1, xrow + (c - 31) * 32);
      compute(c, lA2[c & 1], accNx);
      wait_async0();
      __syncthreads();
    }
    compute(47, lA2[1], accNx);   // last chunk; grid barrier below resyncs

    // ---- gate epilogue (f32 C layout: VGPR v -> row m0+v+8g, col lane&15) ----
    __bf16* __restrict__ hout_bf = hsb + (size_t)(t + 1) * (B_ * H_);
#pragma unroll
    for (int v = 0; v < 8; ++v) {
      const int brow = m0 + v + 8 * g;
      const float r  = sigmoidf_(accR[v] + br);
      const float z  = sigmoidf_(accZ[v] + bz);
      const float n  = tanhf(accNx[v] + bnx + r * (accNh[v] + bnh));
      const float hpv = hp[(size_t)brow * H_ + col];
      const float hv = (1.0f - z) * n + z * hpv;
      hn[(size_t)brow * H_ + col] = hv;
      hout_bf[(size_t)brow * H_ + col] = (__bf16)hv;
    }

    grid_barrier(bar_cnt, bar_gen);
  }
}

// ---------------------------------------------------------------------------
// Output dense: out[b,t,:] = hs[t,b,:](bf16) @ w_dense^T + b_dense.
// ---------------------------------------------------------------------------
#define LDB_S 40
__device__ __forceinline__ v16bf load_b_frag(const __bf16* lB, int n0, int lane) {
  const int g = lane >> 4, r = lane & 15;
  const __bf16* p = lB + (n0 + r) * LDB_S + g * 16;
  v8bf lo = *(const v8bf*)(p);
  v8bf hi = *(const v8bf*)(p + 8);
  v16bf b;
#pragma unroll
  for (int i = 0; i < 8; ++i) { b[i] = lo[i]; b[i + 8] = hi[i]; }
  return b;
}

__global__ __launch_bounds__(512) void dense_out(
    const __bf16* __restrict__ hs,      // [T,B,H] bf16
    const float* __restrict__ w_dense,  // [E, H]
    const float* __restrict__ b_dense,  // [E]
    float* __restrict__ out) {          // [B,T,E]
  __shared__ __align__(16) __bf16 lA[64 * LDA_S];
  __shared__ __align__(16) __bf16 lB[64 * LDB_S];

  const int tid  = threadIdx.x;
  const int lane = tid & 31;
  const int wv   = tid >> 5;
  const int m0   = (wv >> 2) * 16;
  const int n0   = (wv & 3) * 16;
  const int tt   = blockIdx.x;
  const int colbase = blockIdx.y * 64;
  const __bf16* aBase = hs + (size_t)tt * (B_ * H_);

  const int ar = tid >> 3;
  const int ak = (tid & 7) * 4;

  v8f acc = {};
  for (int k0 = 0; k0 < H_; k0 += 32) {
    *(v4bf*)&lA[ar * LDA_S + ak] =
        *(const v4bf*)(aBase + (size_t)ar * H_ + (k0 + ak));
    const float4 w =
        *(const float4*)(w_dense + (size_t)(colbase + ar) * H_ + (k0 + ak));
    *(v4bf*)&lB[ar * LDB_S + ak] = cvt4(w);
    __syncthreads();
    {
      v16bf a = load_a_frag(lA, m0, lane);
      v16bf b = load_b_frag(lB, n0, lane);
      acc = wmma_bf16(a, b, acc);
    }
    __syncthreads();
  }

  const int g   = lane >> 4;
  const int col = colbase + n0 + (lane & 15);
  const float bias = b_dense[col];
#pragma unroll
  for (int v = 0; v < 8; ++v) {
    const int b = m0 + v + 8 * g;
    out[((size_t)b * T_ + tt) * E_ + col] = acc[v] + bias;
  }
}

extern "C" void kernel_launch(void* const* d_in, const int* in_sizes, int n_in,
                              void* d_out, int out_size, void* d_ws, size_t ws_size,
                              hipStream_t stream) {
  const float* x       = (const float*)d_in[0];
  const float* w_ih    = (const float*)d_in[1];
  const float* w_hh    = (const float*)d_in[2];
  const float* b_ih    = (const float*)d_in[3];
  const float* b_hh    = (const float*)d_in[4];
  const float* w_dense = (const float*)d_in[5];
  const float* b_dense = (const float*)d_in[6];
  float* out = (float*)d_out;

  // workspace: h0 f32 | bar(16 u32) | h1 f32 | hsb bf16 [T+1][B][H] | xb bf16
  float* h0 = (float*)d_ws;
  unsigned* bar = (unsigned*)(h0 + B_ * H_);
  float* h1 = (float*)(bar + 16);
  __bf16* hsb = (__bf16*)(h1 + B_ * H_);
  __bf16* xb  = hsb + (size_t)(T_ + 1) * B_ * H_;

  // zero h0 + barrier words (contiguous), and hsb slot 0 (h_{-1} = 0)
  zero_f32<<<(B_ * H_ + 16 + 255) / 256, 256, 0, stream>>>(h0, B_ * H_ + 16);
  zero_f32<<<(B_ * H_ / 2 + 255) / 256, 256, 0, stream>>>((float*)hsb,
                                                          B_ * H_ / 2);
  // pre-convert x to bf16 so recurrence staging is pure async copy
  const int n4 = B_ * T_ * E_ / 4;
  cvt_f32_bf16<<<(n4 + 255) / 256, 256, 0, stream>>>(x, xb, n4);

  gru_persistent<<<dim3(NBLK), 128, 0, stream>>>(xb, w_ih, w_hh, b_ih, b_hh,
                                                 h0, h1, hsb, bar, bar + 1);

  dense_out<<<dim3(T_, E_ / 64), 512, 0, stream>>>(hsb + (size_t)B_ * H_,
                                                   w_dense, b_dense, out);
}
